// BaselineModel_74509092651544
// MI455X (gfx1250) — compile-verified, matrix operand order checked by reading, not run on hardware
//
#include <hip/hip_runtime.h>
#include <hip/hip_bf16.h>

// ---------------------------------------------------------------------------
// MPNN (CLRS-style) on MI455X / gfx1250.
// Strategy: fuse the dense per-slot message MLP (3x [H]x[H,H] per (b,s,r)
// slot, ~155 GFLOP total) into one WMMA kernel that never materializes the
// [B,N,N,H] activations. All matmuls use v_wmma_f32_16x16x32_bf16 with f32
// accumulation. edge_dense is pre-packed ONCE into bf16 A-fragment order so
// the hot loop feeds WMMA straight from coalesced global_load_b128 (134 MB /
// layer, ~L2-resident given 192 MB L2), with global_prefetch for the next
// sender tile. graph_fts stays zero in the reference, so mg == bg[l] and all
// biases fold into the WMMA C initializer.
// ---------------------------------------------------------------------------

#define B_  32
#define N_  128
#define H_  128
#define L_  3
#define E_  65536
#define BN_ 4096
#define NEGV (-1.0e9f)

typedef __attribute__((ext_vector_type(16))) __bf16 v16bf;
typedef __attribute__((ext_vector_type(8)))  float  v8f;
typedef __attribute__((ext_vector_type(2)))  unsigned int u32x2;

__device__ __forceinline__ unsigned short f2bf(float f) {
  unsigned int u = __float_as_uint(f);
  u += 0x7FFFu + ((u >> 16) & 1u);                 // round-to-nearest-even
  return (unsigned short)(u >> 16);
}

// A-fragment (16-bit, 16x32 per K-block) packed offset (in halves) for
// element (row r, k). ISA layout: lane = r + 16*((k'>>3)&1),
// half = (k'&7) + 8*(k'>=16), k' = k & 31.  Frag f stored at f*32*16 halves.
__device__ __forceinline__ int a_half_off(int k, int r) {
  int kp   = k & 31;
  int lane = r + 16 * ((kp >> 3) & 1);
  int half = (kp & 7) + 8 * ((kp >> 4) & 1);
  return ((k >> 5) * 32 + lane) * 16 + half;
}

// ---------------------------------------------------------------------------
// Pack a row-major f32 weight [K,128] into bf16 B-fragment layout:
// out[((kf*8 + ntile)*32 + lane)*16 + half]; inverse of the A mapping with
// col = lane&15, k' = (half&7) + 8*(lane>>4) + 16*(half>>3).
// ---------------------------------------------------------------------------
__global__ void pack_w_kernel(const float* __restrict__ W,
                              unsigned short* __restrict__ out, int nkf) {
  int p = blockIdx.x * 256 + threadIdx.x;
  if (p >= nkf * 4096) return;
  int half = p & 15, lane = (p >> 4) & 31, nt = (p >> 9) & 7, kf = p >> 12;
  int kp = (half & 7) + 8 * (lane >> 4) + 16 * (half >> 3);
  int k  = kf * 32 + kp;
  int nn = nt * 16 + (lane & 15);
  out[p] = f2bf(W[(size_t)k * 128 + nn]);
}

// ---------------------------------------------------------------------------
// One-time repack of edge_dense (f32 [B,N,N,H]) into bf16 A-fragment tiles:
// tile tIdx = (b*128 + s)*8 + rt holds the 16x128 slot rows (receivers
// rt*16..rt*16+15) in WMMA A layout (4 K-frags * 32 lanes * 16 halves).
// Thread per u32 (two consecutive halves = elements (r,k),(r,k+1)).
// ---------------------------------------------------------------------------
__global__ void edge_pack_kernel(const float* __restrict__ edense,
                                 unsigned int* __restrict__ out) {
  int p = blockIdx.x * 256 + threadIdx.x;          // 33,554,432 u32 total
  int tIdx = p >> 10;                              // 1024 u32 per tile
  int within = p & 1023;
  int halfoff = within * 2;
  int kf   = halfoff >> 9;
  int lane = (halfoff >> 4) & 31;
  int half = halfoff & 15;                         // even
  int r  = lane & 15;
  int k  = kf * 32 + (half & 7) + 8 * (lane >> 4) + 16 * (half >> 3);
  int b  = tIdx >> 10;
  int s  = (tIdx >> 3) & 127;
  int rt = tIdx & 7;
  const float* src =
      edense + (((size_t)(b * N_ + s)) * N_ + rt * 16 + r) * H_ + k;
  out[p] = (unsigned int)f2bf(src[0]) | ((unsigned int)f2bf(src[1]) << 16);
}

// ---------------------------------------------------------------------------
// Atom / bond encoders: sum of per-column embeddings.
// ---------------------------------------------------------------------------
__global__ void atom_encode_kernel(const int* __restrict__ x,
                                   const float* __restrict__ emb,
                                   float* __restrict__ out) {
  int i = blockIdx.x, t = threadIdx.x;
  float s = 0.f;
#pragma unroll
  for (int j = 0; j < 9; ++j) {
    int v = x[i * 9 + j];
    s += emb[((size_t)j * 128 + v) * 128 + t];     // AV = 128
  }
  out[(size_t)i * 128 + t] = s;
}

__global__ void bond_encode_kernel(const int* __restrict__ ea,
                                   const float* __restrict__ emb,
                                   float* __restrict__ out) {
  int e = blockIdx.x, t = threadIdx.x;
  float s = 0.f;
#pragma unroll
  for (int j = 0; j < 3; ++j) {
    int v = ea[e * 3 + j];
    s += emb[((size_t)j * 16 + v) * 128 + t];      // BV = 16
  }
  out[(size_t)e * 128 + t] = s;
}

// Scatter edge features + adjacency mask into dense [B,N,N,(H)] buffers.
__global__ void scatter_edges_kernel(const int* __restrict__ eidx,
                                     const float* __restrict__ efts,
                                     float* __restrict__ edense,
                                     unsigned char* __restrict__ adj) {
  int e = blockIdx.x, t = threadIdx.x;
  int a = eidx[e], d = eidx[E_ + e];
  int g = a / N_, sl = a % N_, dl = d % N_;
  size_t slot = ((size_t)(g * N_ + sl)) * N_ + dl;
  atomicAdd(&edense[slot * H_ + t], efts[(size_t)e * H_ + t]);
  if (t == 0) adj[slot] = 1;
}

// ---------------------------------------------------------------------------
// Generic one-wave WMMA GEMM: out[16,128] = act( [A0|A1][16,K] @ W + bias
//                                                (+ addend) ).
// NKF = K/32 (4 or 8).  pW is pre-packed bf16 (B-fragment layout).
// ---------------------------------------------------------------------------
template <int NKF>
__global__ __launch_bounds__(32) void gemm16_kernel(
    const float* __restrict__ A0, const float* __restrict__ A1,
    const unsigned short* __restrict__ pW, const float* __restrict__ bias,
    const float* __restrict__ addend, const int do_relu,
    float* __restrict__ outp) {
  __shared__ unsigned short sA[8 * 32 * 16];
  const int lane = threadIdx.x;
  const int n = lane & 15, hi = lane >> 4;
  const int row0 = blockIdx.x * 16;

  const int nsrc = (NKF > 4) ? 2 : 1;
  for (int sI = 0; sI < nsrc; ++sI) {
    const float* Ap = (sI == 0 ? A0 : A1) + (size_t)row0 * 128;
#pragma unroll
    for (int t = 0; t < 16; ++t) {
      int idx = lane + 32 * t;                     // 512 float4 = 16x128 f32
      int r = idx >> 5, c4 = idx & 31, k = c4 * 4;
      const float4 v = *reinterpret_cast<const float4*>(Ap + (size_t)r * 128 + k);
      u32x2 p;
      p.x = (unsigned int)f2bf(v.x) | ((unsigned int)f2bf(v.y) << 16);
      p.y = (unsigned int)f2bf(v.z) | ((unsigned int)f2bf(v.w) << 16);
      *reinterpret_cast<u32x2*>(&sA[a_half_off(k + sI * 128, r)]) = p;
    }
  }
  asm volatile("s_wait_dscnt 0x0" ::: "memory");   // cross-lane LDS handoff

  v16bf af[NKF];
#pragma unroll
  for (int kf = 0; kf < NKF; ++kf)
    af[kf] = *reinterpret_cast<const v16bf*>(&sA[(kf * 32 + lane) * 16]);

#pragma unroll
  for (int j = 0; j < 8; ++j) {
    int h = 16 * j + n;
    float bj = bias[h];
    v8f c;
#pragma unroll
    for (int e = 0; e < 8; ++e) c[e] = bj;
#pragma unroll
    for (int kf = 0; kf < NKF; ++kf) {
      v16bf bf = *reinterpret_cast<const v16bf*>(
          pW + ((size_t)(kf * 8 + j) * 32 + lane) * 16);
      c = __builtin_amdgcn_wmma_f32_16x16x32_bf16(false, af[kf], false, bf,
                                                  (short)0, c, false, false);
    }
#pragma unroll
    for (int e = 0; e < 8; ++e) {
      int r = row0 + e + 8 * hi;                   // C layout: row = v + 8*hi
      float f = c[e];
      if (addend) f += addend[(size_t)r * 128 + h];
      if (do_relu) f = fmaxf(f, 0.0f);
      outp[(size_t)r * 128 + h] = f;
    }
  }
}

// ---------------------------------------------------------------------------
// Fused message kernel. Grid (8 receiver tiles, 32 graphs), 8 waves/WG.
// Wave w handles senders [16w, 16w+16). Per sender: load the pre-packed bf16
// edge tile straight into A fragments (coalesced global_load_b128), then 3
// chained WMMA GEMM stages (We, Wp1, Wp2) with LDS fragment handoffs, then
// adjacency-masked running max in C-layout registers. Next sender tile is
// prefetched (global_prefetch) while the GEMMs run.
// ---------------------------------------------------------------------------
__global__ __launch_bounds__(256) void mpnn_msgs_kernel(
    const unsigned short* __restrict__ epk,   // packed edge tiles
    const unsigned char* __restrict__ adj,
    const float* __restrict__ m1, const float* __restrict__ m2,
    const float* __restrict__ be, const float* __restrict__ bg,
    const float* __restrict__ bp1, const float* __restrict__ bp2,
    const unsigned short* __restrict__ pWe,
    const unsigned short* __restrict__ pWp1,
    const unsigned short* __restrict__ pWp2,
    float* __restrict__ mmax) {
  extern __shared__ char smem[];
  unsigned short* sW   = (unsigned short*)smem;          // 3*16384 halves
  unsigned short* sA   = sW + 3 * 16384;                 // 8 waves * 2048 halves
  float*          slab = (float*)(sA + 8 * 2048);        // 8 waves * 2048 f32
  unsigned int*  mbits = (unsigned int*)(slab + 8 * 2048); // 128 sender masks

  const int rt = blockIdx.x, b = blockIdx.y;
  const int tid = threadIdx.x;
  const int lane = tid & 31, wv = tid >> 5;
  const int n = lane & 15, hi = lane >> 4;
  const int r0 = rt * 16;

  { // stage the three packed weights into LDS (coalesced u32 copies)
    const unsigned int* s0 = (const unsigned int*)pWe;
    const unsigned int* s1 = (const unsigned int*)pWp1;
    const unsigned int* s2 = (const unsigned int*)pWp2;
    unsigned int* d = (unsigned int*)sW;
    for (int i = tid; i < 8192; i += 256) {
      d[i] = s0[i]; d[8192 + i] = s1[i]; d[16384 + i] = s2[i];
    }
  }
  if (tid < 128) { // per-sender 16-bit adjacency mask over this receiver tile
    unsigned int m = 0;
    const unsigned char* arow = adj + ((size_t)(b * N_ + tid)) * N_ + r0;
#pragma unroll
    for (int r = 0; r < 16; ++r) m |= arow[r] ? (1u << r) : 0u;
    mbits[tid] = m;
  }

  // receiver-side constants held in registers (identical across waves):
  // m2t = m2[r] + be + bg (mg == bg since graph_fts stays zero)
  float m2t[8][8], bp1r[8], bp2r[8];
#pragma unroll
  for (int j = 0; j < 8; ++j) {
    int h = 16 * j + n;
    bp1r[j] = bp1[h];
    bp2r[j] = bp2[h];
    float ebg = be[h] + bg[h];
#pragma unroll
    for (int e = 0; e < 8; ++e) {
      int rr = e + 8 * hi;
      m2t[j][e] = m2[(size_t)(b * N_ + r0 + rr) * 128 + h] + ebg;
    }
  }
  float mx[8][8];
#pragma unroll
  for (int j = 0; j < 8; ++j)
#pragma unroll
    for (int e = 0; e < 8; ++e) mx[j][e] = NEGV;

  __syncthreads();

  unsigned short* sAw = sA + wv * 2048;            // private per-wave slab
  const size_t tile_base = ((size_t)(b * N_ + wv * 16) * 8 + rt) * 2048;

  for (int si = 0; si < 16; ++si) {
    const int s = wv * 16 + si;
    const unsigned short* tile = epk + tile_base + (size_t)si * (8 * 2048);

    // A fragments straight from global (32B/lane, coalesced b128 loads)
    v16bf af[4];
#pragma unroll
    for (int kf = 0; kf < 4; ++kf)
      af[kf] = *reinterpret_cast<const v16bf*>(tile + (kf * 32 + lane) * 16);

    // prefetch next sender's tile (4 KB) while the 3 GEMM stages run
    if (si < 15)
      __builtin_prefetch(tile + 8 * 2048 + lane * 64, 0, 1);

    float m1r[8];
    const float* m1p = m1 + (size_t)(b * N_ + s) * 128;
#pragma unroll
    for (int j = 0; j < 8; ++j) m1r[j] = m1p[16 * j + n];
    const unsigned int mbs = mbits[s];

    // ---- stage 1: t = relu( edge @ We + m1[s] + m2[r] + be + bg ) ----
#pragma unroll
    for (int j = 0; j < 8; ++j) {
      v8f c;
#pragma unroll
      for (int e = 0; e < 8; ++e) c[e] = m1r[j] + m2t[j][e];
#pragma unroll
      for (int kf = 0; kf < 4; ++kf) {
        v16bf bf = *reinterpret_cast<const v16bf*>(
            &sW[((kf * 8 + j) * 32 + lane) * 16]);
        c = __builtin_amdgcn_wmma_f32_16x16x32_bf16(false, af[kf], false, bf,
                                                    (short)0, c, false, false);
      }
#pragma unroll
      for (int e = 0; e < 8; ++e) {   // C layout -> packed A layout in LDS
        float f = fmaxf(c[e], 0.0f);
        int r = e + 8 * hi;
        int off = ((j >> 1) * 32 + r + 16 * (n >> 3)) * 16 + (n & 7) + 8 * (j & 1);
        sAw[off] = f2bf(f);
      }
    }
    asm volatile("s_wait_dscnt 0x0" ::: "memory");
#pragma unroll
    for (int kf = 0; kf < 4; ++kf)
      af[kf] = *reinterpret_cast<const v16bf*>(&sAw[(kf * 32 + lane) * 16]);

    // ---- stage 2: u = relu( t @ Wp1 + bp1 ) ----
#pragma unroll
    for (int j = 0; j < 8; ++j) {
      v8f c;
#pragma unroll
      for (int e = 0; e < 8; ++e) c[e] = bp1r[j];
#pragma unroll
      for (int kf = 0; kf < 4; ++kf) {
        v16bf bf = *reinterpret_cast<const v16bf*>(
            &sW[16384 + ((kf * 8 + j) * 32 + lane) * 16]);
        c = __builtin_amdgcn_wmma_f32_16x16x32_bf16(false, af[kf], false, bf,
                                                    (short)0, c, false, false);
      }
#pragma unroll
      for (int e = 0; e < 8; ++e) {
        float f = fmaxf(c[e], 0.0f);
        int r = e + 8 * hi;
        int off = ((j >> 1) * 32 + r + 16 * (n >> 3)) * 16 + (n & 7) + 8 * (j & 1);
        sAw[off] = f2bf(f);
      }
    }
    asm volatile("s_wait_dscnt 0x0" ::: "memory");
#pragma unroll
    for (int kf = 0; kf < 4; ++kf)
      af[kf] = *reinterpret_cast<const v16bf*>(&sAw[(kf * 32 + lane) * 16]);

    // ---- stage 3: v = u @ Wp2 + bp2 ; masked running max over senders ----
#pragma unroll
    for (int j = 0; j < 8; ++j) {
      v8f c;
#pragma unroll
      for (int e = 0; e < 8; ++e) c[e] = bp2r[j];
#pragma unroll
      for (int kf = 0; kf < 4; ++kf) {
        v16bf bf = *reinterpret_cast<const v16bf*>(
            &sW[32768 + ((kf * 8 + j) * 32 + lane) * 16]);
        c = __builtin_amdgcn_wmma_f32_16x16x32_bf16(false, af[kf], false, bf,
                                                    (short)0, c, false, false);
      }
#pragma unroll
      for (int e = 0; e < 8; ++e) {
        int r = e + 8 * hi;
        float vv = ((mbs >> r) & 1u) ? c[e] : NEGV;
        mx[j][e] = fmaxf(mx[j][e], vv);
      }
    }
  }

  // cross-wave max combine via LDS
  float* sl = slab + wv * 2048;
#pragma unroll
  for (int j = 0; j < 8; ++j)
#pragma unroll
    for (int e = 0; e < 8; ++e)
      sl[(e + 8 * hi) * 128 + 16 * j + n] = mx[j][e];
  __syncthreads();
  for (int u = tid; u < 2048; u += 256) {
    float m = NEGV;
#pragma unroll
    for (int w = 0; w < 8; ++w) m = fmaxf(m, slab[w * 2048 + u]);
    int r = u >> 7, h = u & 127;
    mmax[(size_t)(b * N_ + r0 + r) * 128 + h] = m;
  }
}

// ---------------------------------------------------------------------------
// LayerNorm (per node row of 128)
// ---------------------------------------------------------------------------
__global__ void ln_kernel(const float* __restrict__ hsrc,
                          const float* __restrict__ g,
                          const float* __restrict__ bt,
                          float* __restrict__ dst) {
  int row = blockIdx.x, t = threadIdx.x;
  __shared__ float buf[128];
  float v = hsrc[(size_t)row * 128 + t];
  buf[t] = v;
  __syncthreads();
  for (int off = 64; off; off >>= 1) {
    if (t < off) buf[t] += buf[t + off];
    __syncthreads();
  }
  float mu = buf[0] * (1.0f / 128.0f);
  __syncthreads();
  float d = v - mu;
  buf[t] = d * d;
  __syncthreads();
  for (int off = 64; off; off >>= 1) {
    if (t < off) buf[t] += buf[t + off];
    __syncthreads();
  }
  float var = buf[0] * (1.0f / 128.0f);
  dst[(size_t)row * 128 + t] = d * rsqrtf(var + 1e-5f) * g[t] + bt[t];
}

// Mean pool + 2-layer prediction head (tiny: 2.6 MFLOP total)
__global__ void head_kernel(const float* __restrict__ hidden,
                            const float* __restrict__ Wh1,
                            const float* __restrict__ bh1,
                            const float* __restrict__ Wh2,
                            const float* __restrict__ bh2,
                            float* __restrict__ out) {
  int b = blockIdx.x, t = threadIdx.x;
  __shared__ float ge[128], tb[128];
  float s = 0.f;
  for (int nn = 0; nn < N_; ++nn) s += hidden[((size_t)b * N_ + nn) * 128 + t];
  ge[t] = s * (1.0f / (float)N_);
  __syncthreads();
  float a = bh1[t];
  for (int h = 0; h < 128; ++h) a += ge[h] * Wh1[(size_t)h * 128 + t];
  tb[t] = fmaxf(a, 0.f);
  __syncthreads();
  float o = bh2[t];
  for (int h = 0; h < 128; ++h) o += tb[h] * Wh2[(size_t)h * 128 + t];
  out[(size_t)b * 128 + t] = o;
}

// ---------------------------------------------------------------------------
extern "C" void kernel_launch(void* const* d_in, const int* in_sizes, int n_in,
                              void* d_out, int out_size, void* d_ws, size_t ws_size,
                              hipStream_t stream) {
  const int*   x         = (const int*)d_in[0];
  const int*   edge_attr = (const int*)d_in[1];
  const int*   edge_idx  = (const int*)d_in[2];
  /* d_in[3] batch_idx unused (equal-size graphs) */
  const float* atom_emb  = (const float*)d_in[4];
  const float* bond_emb  = (const float*)d_in[5];
  const float* Wm1 = (const float*)d_in[6];  const float* bm1 = (const float*)d_in[7];
  const float* Wm2 = (const float*)d_in[8];  const float* bm2 = (const float*)d_in[9];
  const float* We  = (const float*)d_in[10]; const float* be  = (const float*)d_in[11];
  /* Wg (d_in[12]) unused: graph_fts stays zero -> mg == bg */
  const float* bg  = (const float*)d_in[13];
  const float* Wp1 = (const float*)d_in[14]; const float* bp1 = (const float*)d_in[15];
  const float* Wp2 = (const float*)d_in[16]; const float* bp2 = (const float*)d_in[17];
  const float* Wo1 = (const float*)d_in[18]; const float* bo1 = (const float*)d_in[19];
  const float* Wo2 = (const float*)d_in[20]; const float* bo2 = (const float*)d_in[21];
  const float* lns = (const float*)d_in[22]; const float* lnb = (const float*)d_in[23];
  const float* Wh1 = (const float*)d_in[24]; const float* bh1 = (const float*)d_in[25];
  const float* Wh2 = (const float*)d_in[26]; const float* bh2 = (const float*)d_in[27];

  // workspace carve-up (~452 MB total)
  char* ws = (char*)d_ws;
  size_t off = 0;
  auto carve = [&](size_t bytes) { char* p = ws + off; off = (off + bytes + 255) & ~(size_t)255; return p; };
  float*          edense   = (float*)carve((size_t)B_ * N_ * N_ * H_ * 4);   // 268 MB
  unsigned int*   epk      = (unsigned int*)carve((size_t)B_ * N_ * N_ * H_ * 2); // 134 MB packed bf16
  unsigned char*  adj      = (unsigned char*)carve((size_t)B_ * N_ * N_);
  float*          node_fts = (float*)carve((size_t)BN_ * H_ * 4);
  float*          edge_fts = (float*)carve((size_t)E_ * H_ * 4);             // 32 MB
  float*          hidden   = (float*)carve((size_t)BN_ * H_ * 4);
  float*          m1buf    = (float*)carve((size_t)BN_ * H_ * 4);
  float*          m2buf    = (float*)carve((size_t)BN_ * H_ * 4);
  float*          zobuf    = (float*)carve((size_t)BN_ * H_ * 4);
  float*          mmax     = (float*)carve((size_t)BN_ * H_ * 4);
  float*          hbuf     = (float*)carve((size_t)BN_ * H_ * 4);
  unsigned short* pWm1     = (unsigned short*)carve(8 * 4096 * 2);
  unsigned short* pWm2     = (unsigned short*)carve(8 * 4096 * 2);
  unsigned short* pWo1     = (unsigned short*)carve(8 * 4096 * 2);
  unsigned short* pWe      = (unsigned short*)carve(4 * 4096 * 2);
  unsigned short* pWp1     = (unsigned short*)carve(4 * 4096 * 2);
  unsigned short* pWp2     = (unsigned short*)carve(4 * 4096 * 2);
  unsigned short* pWo2     = (unsigned short*)carve(4 * 4096 * 2);

  hipMemsetAsync(edense, 0, (size_t)B_ * N_ * N_ * H_ * 4, stream);
  hipMemsetAsync(adj,    0, (size_t)B_ * N_ * N_, stream);
  hipMemsetAsync(hidden, 0, (size_t)BN_ * H_ * 4, stream);

  atom_encode_kernel<<<BN_, 128, 0, stream>>>(x, atom_emb, node_fts);
  bond_encode_kernel<<<E_, 128, 0, stream>>>(edge_attr, bond_emb, edge_fts);
  scatter_edges_kernel<<<E_, 128, 0, stream>>>(edge_idx, edge_fts, edense, adj);
  // one-time repack: f32 dense edge tensor -> bf16 WMMA A-fragment tiles
  edge_pack_kernel<<<131072, 256, 0, stream>>>(edense, epk);

  const size_t msg_smem = 3 * 16384 * 2    // packed We/Wp1/Wp2 (bf16)
                        + 8 * 2048 * 2     // per-wave A-fragment handoff slabs
                        + 8 * 2048 * 4     // per-wave max slabs
                        + 128 * 4;         // sender adjacency bitmasks

  for (int l = 0; l < L_; ++l) {
    pack_w_kernel<<<8 * 16, 256, 0, stream>>>(Wm1 + (size_t)l * 256 * 128, pWm1, 8);
    pack_w_kernel<<<8 * 16, 256, 0, stream>>>(Wm2 + (size_t)l * 256 * 128, pWm2, 8);
    pack_w_kernel<<<8 * 16, 256, 0, stream>>>(Wo1 + (size_t)l * 256 * 128, pWo1, 8);
    pack_w_kernel<<<4 * 16, 256, 0, stream>>>(We  + (size_t)l * 128 * 128, pWe,  4);
    pack_w_kernel<<<4 * 16, 256, 0, stream>>>(Wp1 + (size_t)l * 128 * 128, pWp1, 4);
    pack_w_kernel<<<4 * 16, 256, 0, stream>>>(Wp2 + (size_t)l * 128 * 128, pWp2, 4);
    pack_w_kernel<<<4 * 16, 256, 0, stream>>>(Wo2 + (size_t)l * 128 * 128, pWo2, 4);

    // z = [node_fts | hidden]  ->  m1, m2, zo   ([4096,256] @ [256,128])
    gemm16_kernel<8><<<BN_ / 16, 32, 0, stream>>>(node_fts, hidden, pWm1,
                                                  bm1 + l * H_, nullptr, 0, m1buf);
    gemm16_kernel<8><<<BN_ / 16, 32, 0, stream>>>(node_fts, hidden, pWm2,
                                                  bm2 + l * H_, nullptr, 0, m2buf);
    gemm16_kernel<8><<<BN_ / 16, 32, 0, stream>>>(node_fts, hidden, pWo1,
                                                  bo1 + l * H_, nullptr, 0, zobuf);

    // fused message pipeline + masked max over senders
    mpnn_msgs_kernel<<<dim3(8, B_), 256, msg_smem, stream>>>(
        (const unsigned short*)epk, adj, m1buf, m2buf, be + l * H_, bg + l * H_,
        bp1 + l * H_, bp2 + l * H_, pWe, pWp1, pWp2, mmax);

    // h = relu( zo + mmax @ Wo2 + bo2 ); layernorm -> hidden
    gemm16_kernel<4><<<BN_ / 16, 32, 0, stream>>>(mmax, nullptr, pWo2,
                                                  bo2 + l * H_, zobuf, 1, hbuf);
    ln_kernel<<<BN_, 128, 0, stream>>>(hbuf, lns + l * H_, lnb + l * H_, hidden);
  }

  head_kernel<<<B_, 128, 0, stream>>>(hidden, Wh1, bh1, Wh2, bh2, (float*)d_out);
}
